// RnntJoiner_13726715478537
// MI455X (gfx1250) — matrix operand order repaired and checked by reading, not compiled, not run on hardware
//
#include <hip/hip_runtime.h>

typedef __attribute__((ext_vector_type(16))) _Float16 v16h;
typedef __attribute__((ext_vector_type(8)))  _Float16 v8h;
typedef __attribute__((ext_vector_type(4)))  _Float16 v4h;
typedef __attribute__((ext_vector_type(8)))  float    v8f;
typedef __attribute__((ext_vector_type(4)))  float    v4f;

#define BB_ 4
#define TT_ 256
#define UU_ 64
#define DD_ 512
#define VV_ 1024

#define A_STRIDE 520    // halfs per staged-A row (512 + 8 pad -> conflict-free ds_load_b128)
#define L_STRIDE 1028   // floats per logits row (1024 + 4 pad)

// ---------------------------------------------------------------------------
// Kernel 1: convert W (V x D, f32 row-major) -> f16 row-major in d_ws, and
// bit-copy x_lens / y_lens into the tail of d_out (tuple outputs 2 and 3).
// ---------------------------------------------------------------------------
__global__ void cvt_w_kernel(const float* __restrict__ W, _Float16* __restrict__ Wh,
                             const int* __restrict__ xl, const int* __restrict__ yl,
                             float* __restrict__ out_tail) {
    int idx  = blockIdx.x * blockDim.x + threadIdx.x;   // one float4 each
    int base = idx << 2;
    if (base < VV_ * DD_) {
        v4f w = *(const v4f*)(W + base);
        v4h h;
        h.x = (_Float16)w.x; h.y = (_Float16)w.y;
        h.z = (_Float16)w.z; h.w = (_Float16)w.w;
        *(v4h*)(Wh + base) = h;
    }
    if (blockIdx.x == 0 && threadIdx.x < 2 * BB_) {
        int t = threadIdx.x;
        int v = (t < BB_) ? xl[t] : yl[t - BB_];
        out_tail[t] = __int_as_float(v);
    }
}

// ---------------------------------------------------------------------------
// Kernel 2: fused relu(x+y) -> f16 GEMM (WMMA) -> +bias -> softmax -> probs.
// One workgroup = 16 rows (fixed b,t ; u0..u0+15) x full V=1024.
// 8 waves; wave w owns columns [w*128, w*128+128).
// ---------------------------------------------------------------------------
__global__ __launch_bounds__(256)
void joiner_kernel(const float* __restrict__ x, const float* __restrict__ y,
                   const _Float16* __restrict__ Wh, const float* __restrict__ bias,
                   float* __restrict__ out) {
    __shared__ __align__(32) unsigned char smem[16 * L_STRIDE * 4]; // 65792 B (union A / logits)
    _Float16* Ah = (_Float16*)smem;   // 16 x A_STRIDE halfs  (16640 B)
    float*    Lg = (float*)smem;      // 16 x L_STRIDE floats (65792 B)

    const int tid  = threadIdx.x;
    const int lane = tid & 31;
    const int wave = tid >> 5;

    const int wg = blockIdx.x;        // 4096 = B * (T*U/16)
    const int bb = wg >> 10;
    const int rt = wg & 1023;
    const int t  = rt >> 2;           // U/16 = 4 row-tiles per t
    const int u0 = (rt & 3) << 4;

    const float* xrow  = x + ((size_t)(bb * TT_ + t)) * DD_;
    const float* ybase = y + ((size_t)(bb * UU_ + u0)) * DD_;

    // ---- stage A = f16(relu(x+y)) into LDS, coalesced float4 reads ----
#pragma unroll
    for (int i = 0; i < 8; ++i) {
        int idx = i * 256 + tid;          // 2048 float4 = 16 rows x 128
        int row = idx >> 7;
        int d   = (idx & 127) << 2;
        v4f xv = *(const v4f*)(xrow + d);
        v4f yv = *(const v4f*)(ybase + (size_t)row * DD_ + d);
        v4h h;
        h.x = (_Float16)fmaxf(xv.x + yv.x, 0.0f);
        h.y = (_Float16)fmaxf(xv.y + yv.y, 0.0f);
        h.z = (_Float16)fmaxf(xv.z + yv.z, 0.0f);
        h.w = (_Float16)fmaxf(xv.w + yv.w, 0.0f);
        *(v4h*)(Ah + row * A_STRIDE + d) = h;
    }
    __syncthreads();

    // ---- GEMM: 16 rows x 128 cols per wave, K = 512 in steps of 32 ----
    const int hi  = lane >> 4;        // half-wave select
    const int l15 = lane & 15;
    // A fragment: lane<16 -> row=l15, K k0..k0+15;
    //             lane>=16 -> K k0+16..k0+31
    const _Float16* Arow  = Ah + l15 * A_STRIDE + (hi ? 16 : 0);
    // B fragment: lane<16 -> col n0+l15, K k0..k0+15; lane>=16 -> K k0+16..k0+31
    const _Float16* Wlane = Wh + (size_t)(wave * 128 + l15) * DD_ + (hi ? 16 : 0);

    v8f acc[8] = {};
    for (int k0 = 0; k0 < DD_; k0 += 32) {
        union { v16h v; v8h h[2]; } au;
        au.h[0] = *(const v8h*)(Arow + k0);        // ds_load_b128
        au.h[1] = *(const v8h*)(Arow + k0 + 8);    // ds_load_b128
        v16h a = au.v;
#pragma unroll
        for (int j = 0; j < 8; ++j) {
            v16h bf = *(const v16h*)(Wlane + (size_t)j * 16 * DD_ + k0); // 32B global load
            acc[j] = __builtin_amdgcn_wmma_f32_16x16x32_f16(
                false, a, false, bf, (short)0, acc[j], false, false);
        }
    }
    __syncthreads();   // everyone done reading A-LDS before reuse as logits

    // ---- bias + spill logits tile to LDS ----
#pragma unroll
    for (int j = 0; j < 8; ++j) {
        int col  = wave * 128 + j * 16 + l15;
        float bv = bias[col];
#pragma unroll
        for (int i = 0; i < 8; ++i) {
            int row = i + (hi ? 8 : 0);   // C layout: VGPR i -> rows i / i+8
            Lg[row * L_STRIDE + col] = acc[j][i] + bv;
        }
    }
    __syncthreads();

    // ---- softmax: wave w handles rows 2w, 2w+1; lane covers 4x8 columns ----
    float* outbase = out + ((size_t)((bb * TT_ + t) * UU_ + u0)) * VV_;
    const float LOG2E = 1.4426950408889634f;
#pragma unroll
    for (int rr = 0; rr < 2; ++rr) {
        int r = wave * 2 + rr;
        float* Lr = Lg + r * L_STRIDE;
        float m = -3.402823466e38f;
#pragma unroll
        for (int i = 0; i < 8; ++i) {
            v4f v = *(v4f*)(Lr + lane * 4 + i * 128);
            m = fmaxf(m, fmaxf(fmaxf(v.x, v.y), fmaxf(v.z, v.w)));
        }
#pragma unroll
        for (int off = 16; off >= 1; off >>= 1)
            m = fmaxf(m, __shfl_xor(m, off, 32));
        float s = 0.0f;
#pragma unroll
        for (int i = 0; i < 8; ++i) {
            v4f v = *(v4f*)(Lr + lane * 4 + i * 128);
            v4f e;
            e.x = exp2f((v.x - m) * LOG2E);
            e.y = exp2f((v.y - m) * LOG2E);
            e.z = exp2f((v.z - m) * LOG2E);
            e.w = exp2f((v.w - m) * LOG2E);
            s += (e.x + e.y) + (e.z + e.w);
            *(v4f*)(Lr + lane * 4 + i * 128) = e;
        }
#pragma unroll
        for (int off = 16; off >= 1; off >>= 1)
            s += __shfl_xor(s, off, 32);
        float inv = 1.0f / s;
        float* orow = outbase + (size_t)r * VV_;
#pragma unroll
        for (int i = 0; i < 8; ++i) {
            v4f e = *(v4f*)(Lr + lane * 4 + i * 128);
            v4f o; o.x = e.x * inv; o.y = e.y * inv; o.z = e.z * inv; o.w = e.w * inv;
            *(v4f*)(orow + lane * 4 + i * 128) = o;   // coalesced b128 stores
        }
    }
}

// ---------------------------------------------------------------------------
extern "C" void kernel_launch(void* const* d_in, const int* in_sizes, int n_in,
                              void* d_out, int out_size, void* d_ws, size_t ws_size,
                              hipStream_t stream) {
    const float* x    = (const float*)d_in[0];
    const int*   xl   = (const int*)  d_in[1];
    const float* y    = (const float*)d_in[2];
    const int*   yl   = (const int*)  d_in[3];
    const float* W    = (const float*)d_in[4];
    const float* bias = (const float*)d_in[5];
    float*    out = (float*)d_out;
    _Float16* Wh  = (_Float16*)d_ws;                 // V*D*2 = 1 MB scratch

    // V*D/4 float4 conversions = 131072 threads
    cvt_w_kernel<<<512, 256, 0, stream>>>(W, Wh, xl, yl,
                                          out + (size_t)BB_ * TT_ * UU_ * VV_);
    // B * T * (U/16) = 4096 row-tiles
    joiner_kernel<<<4096, 256, 0, stream>>>(x, y, Wh, bias, out);
}
